// Net_44160853738178
// MI455X (gfx1250) — matrix operand order repaired and checked by reading, not compiled, not run on hardware
//
#include <hip/hip_runtime.h>
#include <hip/hip_bf16.h>
#include <cstdint>

// ---------------------------------------------------------------------------
// DGCNN-style point net on MI455X (gfx1250).
// Heavy GEMMs: v_wmma_f32_16x16x32_f16 (f16 in, f32 accumulate), with the
// packed weight operand staged through LDS via global_load_async_to_lds_b128
// (ASYNCcnt, double buffered) so all 8 waves in a block share one copy.
// ---------------------------------------------------------------------------

typedef __attribute__((ext_vector_type(16))) _Float16 v16h;
typedef __attribute__((ext_vector_type(8)))  _Float16 v8h;
typedef __attribute__((ext_vector_type(8)))  float    v8f;
typedef _Float16 h16;

#define NPTS   65536
#define NBATCH 256
#define MPTS   256
#define CATLD  800   // [x:0..15(5 used)][a:16..207][b:208..399][c:400..591][d:592..783][pad..799]

// ---------------------------------------------------------------------------
// activation helper
// ---------------------------------------------------------------------------
__device__ __forceinline__ float act_apply(float v, int act) {
    if (act == 1) return v >= 0.f ? v : 0.01f * v;                 // leaky
    if (act == 2) {                                                 // leaky(leaky(.))
        v = v >= 0.f ? v : 0.01f * v;
        return v >= 0.f ? v : 0.01f * v;
    }
    return v;
}

// ---------------------------------------------------------------------------
// Weight pack: f32 [fi x fo] row-major -> WMMA B-fragment layout, f16.
// Fragment element (ktile, ntile, lane, i):
//   lane<16 : K = kt*32 + i,       N = nt*16 + lane
//   lane>=16: K = kt*32 + 16 + i,  N = nt*16 + (lane-16)
// mode 0: identity K map (K<fi valid). mode 1: nn1 remap for CAT layout.
// ---------------------------------------------------------------------------
__device__ __forceinline__ int maprow(int k, int mode, int fi) {
    if (mode == 0) return (k < fi) ? k : -1;
    // mode 1: CAT column -> nn1 weight row (773 rows)
    if (k < 5)   return k;            // x
    if (k < 16)  return -1;           // pad
    if (k < 208) return k - 16  + 5;  // a
    if (k < 400) return k - 208 + 197;// b
    if (k < 592) return k - 400 + 389;// c
    if (k < 784) return k - 592 + 581;// d
    return -1;
}

__global__ void k_pack_w(const float* __restrict__ W, int fi, int fo,
                         int Kpad, int Npad, int mode, h16* __restrict__ out) {
    size_t idx = (size_t)blockIdx.x * 256 + threadIdx.x;
    size_t total = (size_t)(Kpad >> 5) * (Npad >> 4) * 512;
    if (idx >= total) return;
    int i    = (int)(idx & 15);
    int lane = (int)((idx >> 4) & 31);
    size_t tile = idx >> 9;
    int ntiles = Npad >> 4;
    int nt = (int)(tile % ntiles);
    int kt = (int)(tile / ntiles);
    int k  = kt * 32 + ((lane & 16) ? (16 + i) : i);
    int n  = nt * 16 + (lane & 15);
    int src = maprow(k, mode, fi);
    float v = (src >= 0 && n < fo) ? W[(size_t)src * fo + n] : 0.f;
    out[idx] = (h16)v;
}

__global__ void k_pack_b(const float* __restrict__ b, int fo, int Npad,
                         float* __restrict__ out) {
    int t = threadIdx.x;
    if (t < Npad) out[t] = (t < fo) ? b[t] : 0.f;
}

// ---------------------------------------------------------------------------
// WMMA GEMM: C[M x Npad] = act(A[M x K](f16) * Bpacked + bias)
// block = 256 threads = 8 waves; wave w -> rows blockIdx.y*128 + w*16,
// cols blockIdx.x*32 (two 16x16 WMMA tiles). K multiple of 32, M mult of 128.
// Packed B k-slice (1024 halves) is async-copied to LDS, double buffered.
// ---------------------------------------------------------------------------
__global__ __launch_bounds__(256)
void k_gemm(const h16* __restrict__ A, int lda,
            const h16* __restrict__ Bp, const float* __restrict__ bias,
            int M, int Npad, int K, int act,
            float* __restrict__ Cf, int ldcf,
            h16* __restrict__ Ch, int ldch) {
    __shared__ __attribute__((aligned(32))) h16 Bs[2][1024];

    const int tid  = threadIdx.x;
    const int wave = tid >> 5;
    const int lane = tid & 31;
    const int m0 = blockIdx.y * 128 + wave * 16;
    const int n0 = blockIdx.x * 32;
    const int row = m0 + (lane & 15);
    const int kh  = (lane >> 4) * 16;          // K sub-offset per A layout
    const int ntiles = Npad >> 4;
    const size_t nt0 = (size_t)(n0 >> 4);

    // Stage one 32-deep K slice of packed B (2 n-tiles = 2048 bytes) into
    // LDS buffer `buf`: threads 0..127 each move 16 bytes asynchronously.
    auto stageB = [&](int kk, int buf) {
        if (tid < 128) {
            const h16* src = Bp + ((size_t)(kk >> 5) * ntiles + nt0) * 512 + tid * 8;
            unsigned lds_off = (unsigned)(uintptr_t)(&Bs[buf][tid * 8]);
            asm volatile("global_load_async_to_lds_b128 %0, %1, off"
                         :: "v"(lds_off),
                            "v"((unsigned long long)(uintptr_t)src)
                         : "memory");
        }
    };

    v8f c0 = {}; v8f c1 = {};
    const h16* arow = A + (size_t)row * lda + kh;
    const int T = K >> 5;

    stageB(0, 0);
    for (int i = 0; i < T; i++) {
        // buffer i resident for the whole block; previous buffer fully read
        asm volatile("s_wait_asynccnt 0x0" ::: "memory");
        __syncthreads();
        if (i + 1 < T) stageB((i + 1) << 5, (i + 1) & 1);

        const int kk = i << 5;
        v8h alo = *(const v8h*)(arow + kk);
        v8h ahi = *(const v8h*)(arow + kk + 8);
        v16h a;
        #pragma unroll
        for (int e = 0; e < 8; e++) { a[e] = alo[e]; a[8 + e] = ahi[e]; }

        const h16* bbuf = &Bs[i & 1][0];
        v16h b0 = *(const v16h*)(bbuf + lane * 16);
        v16h b1 = *(const v16h*)(bbuf + 512 + lane * 16);
        c0 = __builtin_amdgcn_wmma_f32_16x16x32_f16(false, a, false, b0,
                                                    (short)0, c0, false, false);
        c1 = __builtin_amdgcn_wmma_f32_16x16x32_f16(false, a, false, b1,
                                                    (short)0, c1, false, false);
    }

    // epilogue: C VGPR r holds row m0+r (lanes 0-15) / m0+8+r (lanes 16-31)
    const int rsel = (lane & 16) ? 8 : 0;
    #pragma unroll
    for (int t = 0; t < 2; t++) {
        v8f cc = t ? c1 : c0;
        int col = n0 + t * 16 + (lane & 15);
        float bs = bias[col];
        #pragma unroll
        for (int r = 0; r < 8; r++) {
            int orow = m0 + r + rsel;
            float v = act_apply(cc[r] + bs, act);
            if (Cf) Cf[(size_t)orow * ldcf + col] = v;
            if (Ch) Ch[(size_t)orow * ldch + col] = (h16)v;
        }
    }
}

// ---------------------------------------------------------------------------
// x (N x 5 f32) -> X32 (N x 32 f16, zero padded)
// ---------------------------------------------------------------------------
__global__ void k_prep_x(const float* __restrict__ x, h16* __restrict__ X32) {
    int idx = blockIdx.x * 256 + threadIdx.x;
    int p = idx >> 5, c = idx & 31;
    X32[idx] = (c < 5) ? (h16)x[(size_t)p * 5 + c] : (h16)0.f;
}

// ---------------------------------------------------------------------------
// mask = (clean2 output > 0); write masked x into CAT[:,0:5] (f16) and POS
// (f32), zero CAT pad columns [5,16) and [784,800).
// ---------------------------------------------------------------------------
__global__ __launch_bounds__(256)
void k_mask(const h16* __restrict__ H96b, const float* __restrict__ w,
            const float* __restrict__ b, const float* __restrict__ x,
            h16* __restrict__ CAT, float* __restrict__ POS) {
    int p = blockIdx.x * 256 + threadIdx.x;
    const h16* h = H96b + (size_t)p * 96;
    float s = b[0];
    #pragma unroll 8
    for (int k = 0; k < 96; k++) s += (float)h[k] * w[k];
    float m = (s > 0.f) ? 1.f : 0.f;      // sigmoid(s) > 0.5  <=>  s > 0
    h16* crow = CAT + (size_t)p * CATLD;
    #pragma unroll
    for (int c = 0; c < 5; c++) {
        float xv = x[(size_t)p * 5 + c] * m;
        crow[c] = (h16)xv;
        if (c < 3) POS[(size_t)p * 3 + c] = xv;
    }
    #pragma unroll
    for (int c = 5; c < 16; c++) crow[c] = (h16)0.f;
    #pragma unroll
    for (int c = 784; c < 800; c++) crow[c] = (h16)0.f;
}

// ---------------------------------------------------------------------------
// per-batch KNN (k=4, self excluded, strict-< keeps earliest index on ties)
// ---------------------------------------------------------------------------
__global__ __launch_bounds__(256)
void k_knn(const float* __restrict__ POS, int* __restrict__ NBR) {
    __shared__ float sx[MPTS], sy[MPTS], sz[MPTS];
    int bb = blockIdx.x, i = threadIdx.x;
    int gi = bb * MPTS + i;
    sx[i] = POS[(size_t)gi * 3 + 0];
    sy[i] = POS[(size_t)gi * 3 + 1];
    sz[i] = POS[(size_t)gi * 3 + 2];
    __syncthreads();
    float xi = sx[i], yi = sy[i], zi = sz[i];
    float d0 = 3.4e38f, d1 = 3.4e38f, d2 = 3.4e38f, d3 = 3.4e38f;
    int   i0 = 0, i1 = 0, i2 = 0, i3 = 0;
    for (int j = 0; j < MPTS; j++) {
        if (j == i) continue;
        float dx = sx[j] - xi, dy = sy[j] - yi, dz = sz[j] - zi;
        float d = dx * dx + dy * dy + dz * dz;
        if (d < d3) {
            if (d < d0)      { d3=d2; i3=i2; d2=d1; i2=i1; d1=d0; i1=i0; d0=d; i0=j; }
            else if (d < d1) { d3=d2; i3=i2; d2=d1; i2=i1; d1=d;  i1=j; }
            else if (d < d2) { d3=d2; i3=i2; d2=d;  i2=j; }
            else             { d3=d;  i3=j; }
        }
    }
    NBR[(size_t)gi * 4 + 0] = bb * MPTS + i0;
    NBR[(size_t)gi * 4 + 1] = bb * MPTS + i1;
    NBR[(size_t)gi * 4 + 2] = bb * MPTS + i2;
    NBR[(size_t)gi * 4 + 3] = bb * MPTS + i3;
}

// ---------------------------------------------------------------------------
// Build edge features for one chunk of 16384 points (65536 edge rows):
// E[e, 0:Kf] = xi,  E[e, Kf:2Kf] = xj - xi,  rest 0.
// ---------------------------------------------------------------------------
__global__ void k_edge_gather(const h16* __restrict__ F, int colOff, int Kf,
                              const int* __restrict__ NBR, int pointBase,
                              h16* __restrict__ E, int ldE) {
    size_t idx = (size_t)blockIdx.x * 256 + threadIdx.x;
    int e = (int)(idx / ldE);
    int c = (int)(idx % ldE);
    int p = pointBase + (e >> 2);
    h16 v;
    if (c < Kf) {
        v = F[(size_t)p * CATLD + colOff + c];
    } else if (c < 2 * Kf) {
        int cc = c - Kf;
        int j = NBR[(size_t)p * 4 + (e & 3)];
        v = (h16)((float)F[(size_t)j * CATLD + colOff + cc] -
                  (float)F[(size_t)p * CATLD + colOff + cc]);
    } else {
        v = (h16)0.f;
    }
    E[(size_t)e * ldE + c] = v;
}

// ---------------------------------------------------------------------------
// Sum 4 neighbor rows -> CAT output slice (+ POS for next KNN round)
// ---------------------------------------------------------------------------
__global__ void k_nbr_sum(const float* __restrict__ EO, int pointBase,
                          h16* __restrict__ CAT, int outOff,
                          float* __restrict__ POS, int writePos) {
    size_t idx = (size_t)blockIdx.x * 256 + threadIdx.x;
    int local = (int)(idx / 192);
    int c     = (int)(idx % 192);
    int p = pointBase + local;
    const float* e = EO + (size_t)(local * 4) * 192 + c;
    float s = e[0] + e[192] + e[384] + e[576];
    CAT[(size_t)p * CATLD + outOff + c] = (h16)s;
    if (writePos && c < 3) POS[(size_t)p * 3 + c] = s;
}

// ---------------------------------------------------------------------------
// Head: per-batch segment max/min/sum/mean over 256 pts, leaky, nn3, leaky,
// nn4, tanh on first two outputs.
// ---------------------------------------------------------------------------
__global__ __launch_bounds__(256)
void k_head(const float* __restrict__ H, const float* __restrict__ W3,
            const float* __restrict__ b3, const float* __restrict__ W4,
            const float* __restrict__ b4, float* __restrict__ out) {
    __shared__ float g[768];
    __shared__ float s96[96];
    int bb = blockIdx.x, t = threadIdx.x;
    if (t < 192) {
        float mx = -3.4e38f, mn = 3.4e38f, sm = 0.f;
        const float* col = H + (size_t)(bb * MPTS) * 192 + t;
        for (int p = 0; p < MPTS; p++) {
            float v = col[(size_t)p * 192];
            mx = fmaxf(mx, v); mn = fminf(mn, v); sm += v;
        }
        g[t]       = act_apply(mx, 1);
        g[192 + t] = act_apply(mn, 1);
        g[384 + t] = act_apply(sm, 1);
        g[576 + t] = act_apply(sm * (1.f / MPTS), 1);
    }
    __syncthreads();
    if (t < 96) {
        float acc = b3[t];
        for (int k = 0; k < 768; k++) acc += g[k] * W3[(size_t)k * 96 + t];
        s96[t] = act_apply(acc, 1);
    }
    __syncthreads();
    if (t < 3) {
        float acc = b4[t];
        for (int k = 0; k < 96; k++) acc += s96[k] * W4[(size_t)k * 3 + t];
        if (t < 2) acc = tanhf(acc);
        out[(size_t)bb * 3 + t] = acc;
    }
}

// ---------------------------------------------------------------------------
// Host orchestration
// ---------------------------------------------------------------------------
extern "C" void kernel_launch(void* const* d_in, const int* in_sizes, int n_in,
                              void* d_out, int out_size, void* d_ws, size_t ws_size,
                              hipStream_t stream) {
    (void)in_sizes; (void)n_in; (void)out_size; (void)ws_size;
    const float* x = (const float*)d_in[0];
    // d_in[1] = batch (contiguous 256-point segments by construction; unused)
    auto PW = [&](int pi) { return (const float*)d_in[2 + 2 * pi]; };
    auto PB = [&](int pi) { return (const float*)d_in[3 + 2 * pi]; };
    // param order: 0 clean1_l1, 1 clean1_l2, 2 clean2_l1, 3 clean2_l2,
    //              4..11 conv{1..4}_{l1,l2}, 12 nn1, 13 nn2, 14 nn3, 15 nn4

    size_t off = 0;
    auto carve = [&](size_t bytes) -> void* {
        void* p = (char*)d_ws + off;
        off += (bytes + 255) & ~(size_t)255;
        return p;
    };
    h16*   X32  = (h16*)  carve((size_t)NPTS * 32  * 2);
    h16*   H96  = (h16*)  carve((size_t)NPTS * 96  * 2);
    h16*   H192 = (h16*)  carve((size_t)NPTS * 192 * 2);
    h16*   H96b = (h16*)  carve((size_t)NPTS * 96  * 2);
    h16*   CAT  = (h16*)  carve((size_t)NPTS * CATLD * 2);
    float* POS  = (float*)carve((size_t)NPTS * 3   * 4);
    int*   NBR  = (int*)  carve((size_t)NPTS * 4   * 4);
    h16*   E    = (h16*)  carve((size_t)65536 * 384 * 2);
    h16*   EH   = (h16*)  carve((size_t)65536 * 256 * 2);
    float* EO   = (float*)carve((size_t)65536 * 192 * 4);
    h16*   NH   = (h16*)  carve((size_t)NPTS * 256 * 2);
    float* HFIN = (float*)carve((size_t)NPTS * 192 * 4);
    h16*   WP1  = (h16*)  carve(512 * 1024);
    h16*   WP2  = (h16*)  carve(512 * 1024);
    float* BP1  = (float*)carve(256 * 4);
    float* BP2  = (float*)carve(256 * 4);

    auto pack = [&](const float* W, int fi, int fo, int Kpad, int Npad, int mode, h16* dst) {
        size_t total = (size_t)(Kpad >> 5) * (Npad >> 4) * 512;
        k_pack_w<<<dim3((unsigned)((total + 255) / 256)), 256, 0, stream>>>(
            W, fi, fo, Kpad, Npad, mode, dst);
    };
    auto packb = [&](const float* b, int fo, int Npad, float* dst) {
        k_pack_b<<<1, 256, 0, stream>>>(b, fo, Npad, dst);
    };
    auto gemm = [&](const h16* A, int lda, const h16* Bp, const float* bias,
                    int M, int Npad, int K, int act,
                    float* Cf, int ldcf, h16* Ch, int ldch) {
        dim3 grid((unsigned)(Npad / 32), (unsigned)(M / 128));
        k_gemm<<<grid, 256, 0, stream>>>(A, lda, Bp, bias, M, Npad, K, act,
                                         Cf, ldcf, Ch, ldch);
    };

    // ---- clean stage -----------------------------------------------------
    k_prep_x<<<(NPTS * 32) / 256, 256, 0, stream>>>(x, X32);

    pack(PW(0), 5, 96, 32, 96, 0, WP1);  packb(PB(0), 96, 96, BP1);
    gemm(X32, 32, WP1, BP1, NPTS, 96, 32, /*leaky*/1, nullptr, 0, H96, 96);

    pack(PW(1), 96, 192, 96, 192, 0, WP1);  packb(PB(1), 192, 192, BP1);
    gemm(H96, 96, WP1, BP1, NPTS, 192, 96, /*double leaky*/2, nullptr, 0, H192, 192);

    pack(PW(2), 192, 96, 192, 96, 0, WP1);  packb(PB(2), 96, 96, BP1);
    gemm(H192, 192, WP1, BP1, NPTS, 96, 192, 1, nullptr, 0, H96b, 96);

    k_mask<<<NPTS / 256, 256, 0, stream>>>(H96b, PW(3), PB(3), x, CAT, POS);

    // ---- four edge convs -------------------------------------------------
    for (int cv = 0; cv < 4; cv++) {
        const int Kf     = (cv == 0) ? 5  : 192;
        const int KpadE  = (cv == 0) ? 32 : 384;
        const int fi1    = 2 * Kf;
        const int colOff = (cv == 0) ? 0 : (16 + 192 * (cv - 1));
        const int outOff = 16 + 192 * cv;
        const float* W1 = PW(4 + 2 * cv); const float* b1 = PB(4 + 2 * cv);
        const float* W2 = PW(5 + 2 * cv); const float* b2 = PB(5 + 2 * cv);

        k_knn<<<NBATCH, 256, 0, stream>>>(POS, NBR);

        pack(W1, fi1, 252, KpadE, 256, 0, WP1);  packb(b1, 252, 256, BP1);
        pack(W2, 252, 192, 256,  192, 0, WP2);   packb(b2, 192, 192, BP2);

        for (int ch = 0; ch < 4; ch++) {
            int base = ch * 16384;                       // 16384 pts = 65536 edges
            k_edge_gather<<<(65536u * KpadE) / 256, 256, 0, stream>>>(
                CAT, colOff, Kf, NBR, base, E, KpadE);
            gemm(E, KpadE, WP1, BP1, 65536, 256, KpadE, 1, nullptr, 0, EH, 256);
            gemm(EH, 256,  WP2, BP2, 65536, 192, 256,   1, EO, 192, nullptr, 0);
            k_nbr_sum<<<(16384u * 192) / 256, 256, 0, stream>>>(
                EO, base, CAT, outOff, POS, (cv < 3) ? 1 : 0);
        }
    }

    // ---- nn1 / nn2 -------------------------------------------------------
    pack(PW(12), 773, 252, 800, 256, /*CAT remap*/1, WP1);  packb(PB(12), 252, 256, BP1);
    gemm(CAT, CATLD, WP1, BP1, NPTS, 256, 800, 1, nullptr, 0, NH, 256);

    pack(PW(13), 252, 192, 256, 192, 0, WP2);  packb(PB(13), 192, 192, BP2);
    gemm(NH, 256, WP2, BP2, NPTS, 192, 256, /*no act*/0, HFIN, 192, nullptr, 0);

    // ---- head ------------------------------------------------------------
    k_head<<<NBATCH, 256, 0, stream>>>(HFIN, PW(14), PB(14), PW(15), PB(15),
                                       (float*)d_out);
}
